// FKAMHV_17274358464788
// MI455X (gfx1250) — matrix operand order repaired and checked by reading, not compiled, not run on hardware
//
#include <hip/hip_runtime.h>
#include <hip/hip_bf16.h>
#include <cstddef>

// ---------------- problem dims ----------------
#define MI_N   2048
#define DIS_N  1024
#define DIN_N  3072
#define HID_N  256
#define L_N    15
#define NG_N   8
#define H1_N   512
#define OUT_N  64
#define JK_D   (L_N * HID_N)   // 3840
#define INV_DENOM 1.75f        // 1/(4/7)
#define LN_EPS 1e-5f

// ---------------- types ----------------
typedef __bf16 bhalf;
typedef bhalf  v16bf __attribute__((ext_vector_type(16)));
typedef float  v8f   __attribute__((ext_vector_type(8)));
typedef float  f32x4 __attribute__((ext_vector_type(4)));
typedef unsigned int u32x2 __attribute__((ext_vector_type(2)));
typedef unsigned int u32x4 __attribute__((ext_vector_type(4)));

union Frag { v16bf v; u32x4 q[2]; };

__device__ __forceinline__ unsigned pack2(float a, float b) {
  union { bhalf h[2]; unsigned u; } t;
  t.h[0] = (bhalf)a; t.h[1] = (bhalf)b;
  return t.u;
}
__device__ __forceinline__ float sigmoidf_(float x) { return 1.f / (1.f + __expf(-x)); }
__device__ __forceinline__ float siluf_(float x)    { return x * sigmoidf_(x); }

// ---------------- WMMA GEMM ----------------
// C[M,N] = op_a(A)[M,K] @ B' + (bias) (+C if ACC) then ACT.
//   AOP: 0 = A as-is (fp32, lda), 1 = silu(A), 2 = RBF basis from xn (lda=DIN, K=DIN*NG)
//   BKN: 0 = B stored [N,K] row-major (x @ W.T case), 1 = B stored [K,N] (G @ t case)
//   ACT: 0 = none, 1 = relu
// Block: 256 threads = 8 waves (4 M x 2 N), tile 128x64, K-step 32.
#define BM 128
#define BN 64
#define BK 32
#define LDA_S 40
#define LDB_S 40

template <int AOP, int BKN, int ACC, int BIAS, int ACT>
__global__ __launch_bounds__(256)
void gemm_wmma(const float* __restrict__ A, const float* __restrict__ B,
               const float* __restrict__ bias, const float* __restrict__ grid8,
               float* __restrict__ C, int M, int N, int K,
               int lda, int ldb, int ldc)
{
  __shared__ __align__(16) bhalf As[BM * LDA_S];
  __shared__ __align__(16) bhalf Bs[BN * LDB_S];

  const int tid  = threadIdx.x;
  const int lane = tid & 31;
  const int wid  = tid >> 5;
  const int wm   = wid >> 1;          // 0..3
  const int wn   = wid & 1;           // 0..1
  const int m0   = blockIdx.y * BM;
  const int n0   = blockIdx.x * BN;

  v8f acc[2][2] = {};

  for (int k0 = 0; k0 < K; k0 += BK) {
    // ---- stage A tile: 128x32, 1024 float4-chunks/4 = 4 chunks per thread
#pragma unroll
    for (int c = 0; c < 4; ++c) {
      const int chunk = tid + 256 * c;
      const int row   = chunk >> 3;          // 0..127
      const int kc    = (chunk & 7) * 4;     // 0..28
      float a0, a1, a2, a3;
      if constexpr (AOP == 2) {
        // RBF basis: global K index kg = k0+kc+i ; d = kg>>3 ; g = kg&7
        const int kg = k0 + kc;              // 4 consecutive k share d
        const float xv = A[(size_t)(m0 + row) * lda + (kg >> 3)];
        const f32x4 gv = *(const f32x4*)(grid8 + (kg & 7));
        float t0 = (xv - gv[0]) * INV_DENOM;
        float t1 = (xv - gv[1]) * INV_DENOM;
        float t2 = (xv - gv[2]) * INV_DENOM;
        float t3 = (xv - gv[3]) * INV_DENOM;
        a0 = __expf(-t0 * t0); a1 = __expf(-t1 * t1);
        a2 = __expf(-t2 * t2); a3 = __expf(-t3 * t3);
      } else {
        const f32x4 v = *(const f32x4*)(A + (size_t)(m0 + row) * lda + k0 + kc);
        if constexpr (AOP == 1) {
          a0 = siluf_(v[0]); a1 = siluf_(v[1]); a2 = siluf_(v[2]); a3 = siluf_(v[3]);
        } else {
          a0 = v[0]; a1 = v[1]; a2 = v[2]; a3 = v[3];
        }
      }
      u32x2 w; w[0] = pack2(a0, a1); w[1] = pack2(a2, a3);
      *(u32x2*)(void*)&As[(size_t)row * LDA_S + kc] = w;
    }
    // ---- stage B tile: 64x32 (always stored [n][k] in LDS)
    if constexpr (BKN == 0) {
#pragma unroll
      for (int c = 0; c < 2; ++c) {
        const int chunk = tid + 256 * c;
        const int row   = chunk >> 3;        // n: 0..63
        const int kc    = (chunk & 7) * 4;
        const f32x4 v = *(const f32x4*)(B + (size_t)(n0 + row) * ldb + k0 + kc);
        u32x2 w; w[0] = pack2(v[0], v[1]); w[1] = pack2(v[2], v[3]);
        *(u32x2*)(void*)&Bs[(size_t)row * LDB_S + kc] = w;
      }
    } else {
#pragma unroll
      for (int c = 0; c < 2; ++c) {
        const int chunk = tid + 256 * c;
        const int kr = chunk >> 4;           // k: 0..31
        const int nc = (chunk & 15) * 4;     // n: 0..60
        const f32x4 v = *(const f32x4*)(B + (size_t)(k0 + kr) * ldb + n0 + nc);
#pragma unroll
        for (int i = 0; i < 4; ++i)
          Bs[(size_t)(nc + i) * LDB_S + kr] = (bhalf)v[i];
      }
    }
    __syncthreads();

    // ---- load fragments (ISA 7.12.2 layouts) and WMMA
    Frag fa[2], fb[2];
#pragma unroll
    for (int fm = 0; fm < 2; ++fm) {
      const int row = wm * 32 + fm * 16 + (lane & 15);
      const int kh  = (lane >> 4) * 8;       // A: K halves {kh..kh+7, kh+16..kh+23}
      fa[fm].q[0] = *(const u32x4*)(void*)&As[(size_t)row * LDA_S + kh];
      fa[fm].q[1] = *(const u32x4*)(void*)&As[(size_t)row * LDA_S + kh + 16];
    }
#pragma unroll
    for (int fn = 0; fn < 2; ++fn) {
      const int nn = wn * 32 + fn * 16 + (lane & 15);
      const int kb = (lane >> 4) * 8;        // B: K halves {kb..kb+7, kb+16..kb+23}
      fb[fn].q[0] = *(const u32x4*)(void*)&Bs[(size_t)nn * LDB_S + kb];
      fb[fn].q[1] = *(const u32x4*)(void*)&Bs[(size_t)nn * LDB_S + kb + 16];
    }
#pragma unroll
    for (int fm = 0; fm < 2; ++fm)
#pragma unroll
      for (int fn = 0; fn < 2; ++fn)
        acc[fm][fn] = __builtin_amdgcn_wmma_f32_16x16x32_bf16(
            false, fa[fm].v, false, fb[fn].v, (short)0, acc[fm][fn], false, false);
    __syncthreads();
  }

  // ---- epilogue: C layout = VGPR r -> M=r (lanes 0-15) / M=r+8 (lanes 16-31), N=lane&15
#pragma unroll
  for (int fm = 0; fm < 2; ++fm) {
#pragma unroll
    for (int fn = 0; fn < 2; ++fn) {
      const int nn = n0 + wn * 32 + fn * 16 + (lane & 15);
#pragma unroll
      for (int r = 0; r < 8; ++r) {
        const int mm = m0 + wm * 32 + fm * 16 + r + ((lane >> 4) << 3);
        float v = acc[fm][fn][r];
        if constexpr (BIAS) v += bias[nn];
        if constexpr (ACC)  v += C[(size_t)mm * ldc + nn];
        if constexpr (ACT == 1) v = fmaxf(v, 0.f);
        C[(size_t)mm * ldc + nn] = v;
      }
    }
  }
}

// ---------------- LayerNorm over rows ----------------
__global__ __launch_bounds__(256)
void ln_rows(const float* __restrict__ x, const float* __restrict__ g,
             const float* __restrict__ b, float* __restrict__ y, int D)
{
  const int row = blockIdx.x, tid = threadIdx.x;
  const float* xr = x + (size_t)row * D;
  float s = 0.f, s2 = 0.f;
  for (int j = tid; j < D; j += 256) { float v = xr[j]; s += v; s2 += v * v; }
  __shared__ float r1[256], r2[256];
  r1[tid] = s; r2[tid] = s2; __syncthreads();
  for (int st = 128; st > 0; st >>= 1) {
    if (tid < st) { r1[tid] += r1[tid + st]; r2[tid] += r2[tid + st]; }
    __syncthreads();
  }
  const float mean = r1[0] / D;
  const float var  = r2[0] / D - mean * mean;
  const float rstd = rsqrtf(var + LN_EPS);
  float* yr = y + (size_t)row * D;
  for (int j = tid; j < D; j += 256)
    yr[j] = (xr[j] - mean) * rstd * g[j] + b[j];
}

// ---------------- per-layer head mixing: WeffT[o][i] = sum_h a[h]*W[h][i][o]/H ----------------
__global__ __launch_bounds__(256)
void weff_kernel(const float* __restrict__ W, const float* __restrict__ attn,
                 float* __restrict__ WeffT)
{
  const int idx = blockIdx.x * 256 + threadIdx.x;  // 65536
  const int o = idx & 255, i = idx >> 8;
  float s = 0.f;
#pragma unroll
  for (int h = 0; h < 8; ++h)
    s += attn[h] * W[(size_t)h * HID_N * HID_N + (size_t)i * HID_N + o];
  WeffT[(size_t)o * HID_N + i] = s * 0.125f;
}

// ---------------- gate*sigmoid -> LN -> leaky(0.25), write xc and jk slice ----------------
__global__ __launch_bounds__(256)
void gate_ln_leaky(const float* __restrict__ outlin, const float* __restrict__ glog,
                   const float* __restrict__ g, const float* __restrict__ b,
                   float* __restrict__ xc, float* __restrict__ jk)
{
  const int row = blockIdx.x, j = threadIdx.x;
  float v = outlin[(size_t)row * HID_N + j] * sigmoidf_(glog[(size_t)row * HID_N + j]);
  __shared__ float r1[256], r2[256];
  r1[j] = v; r2[j] = v * v; __syncthreads();
  for (int st = 128; st > 0; st >>= 1) {
    if (j < st) { r1[j] += r1[j + st]; r2[j] += r2[j + st]; }
    __syncthreads();
  }
  const float mean = r1[0] * (1.f / HID_N);
  const float var  = r2[0] * (1.f / HID_N) - mean * mean;
  float y = (v - mean) * rsqrtf(var + LN_EPS) * g[j] + b[j];
  y = (y >= 0.f) ? y : 0.25f * y;
  xc[(size_t)row * HID_N + j] = y;
  jk[(size_t)row * JK_D + j]  = y;
}

// ---------------- final blend ----------------
__global__ __launch_bounds__(256)
void score_kernel(float* __restrict__ score, const float* __restrict__ recon,
                  const float* __restrict__ rh, int n)
{
  const int i = blockIdx.x * 256 + threadIdx.x;
  if (i < n)
    score[i] = 0.7f * sigmoidf_(recon[i]) + 0.3f * sigmoidf_(rh[i]);
}

// ---------------- launcher ----------------
extern "C" void kernel_launch(void* const* d_in, const int* in_sizes, int n_in,
                              void* d_out, int out_size, void* d_ws, size_t ws_size,
                              hipStream_t stream)
{
  (void)in_sizes; (void)n_in; (void)out_size; (void)ws_size;

  // setup_inputs() dict order
  const float* concat_mi  = (const float*)d_in[0];
  const float* concat_dis = (const float*)d_in[1];
  const float* G_mi       = (const float*)d_in[2];
  const float* G_dis      = (const float*)d_in[3];
  // d_in[4]=AT, d_in[5]=A : unused by forward
  const float* rbf_grid   = (const float*)d_in[6];
  // mi block: 7..22
  const float* mi_kln_g = (const float*)d_in[7];
  const float* mi_kln_b = (const float*)d_in[8];
  const float* mi_kWs   = (const float*)d_in[9];
  const float* mi_kWb   = (const float*)d_in[10];
  const float* mi_kbb   = (const float*)d_in[11];
  const float* mi_W     = (const float*)d_in[12];
  const float* mi_attn  = (const float*)d_in[13];
  const float* mi_bias  = (const float*)d_in[14];
  const float* mi_rW1   = (const float*)d_in[15];
  const float* mi_rb1   = (const float*)d_in[16];
  const float* mi_rW2   = (const float*)d_in[17];
  const float* mi_rb2   = (const float*)d_in[18];
  const float* mi_lng   = (const float*)d_in[19];
  const float* mi_lnb   = (const float*)d_in[20];
  const float* mi_jkW   = (const float*)d_in[21];
  const float* mi_jkb   = (const float*)d_in[22];
  // dis block: 23..38
  const float* dis_kln_g = (const float*)d_in[23];
  const float* dis_kln_b = (const float*)d_in[24];
  const float* dis_kWs   = (const float*)d_in[25];
  const float* dis_kWb   = (const float*)d_in[26];
  const float* dis_kbb   = (const float*)d_in[27];
  const float* dis_W     = (const float*)d_in[28];
  const float* dis_attn  = (const float*)d_in[29];
  const float* dis_bias  = (const float*)d_in[30];
  const float* dis_rW1   = (const float*)d_in[31];
  const float* dis_rb1   = (const float*)d_in[32];
  const float* dis_rW2   = (const float*)d_in[33];
  const float* dis_rb2   = (const float*)d_in[34];
  const float* dis_lng   = (const float*)d_in[35];
  const float* dis_lnb   = (const float*)d_in[36];
  const float* dis_jkW   = (const float*)d_in[37];
  const float* dis_jkb   = (const float*)d_in[38];
  // heads
  const float* lx_W1 = (const float*)d_in[39]; const float* lx_b1 = (const float*)d_in[40];
  const float* lx_W2 = (const float*)d_in[41]; const float* lx_b2 = (const float*)d_in[42];
  const float* lx_W3 = (const float*)d_in[43]; const float* lx_b3 = (const float*)d_in[44];
  const float* ly_W1 = (const float*)d_in[45]; const float* ly_b1 = (const float*)d_in[46];
  const float* ly_W2 = (const float*)d_in[47]; const float* ly_b2 = (const float*)d_in[48];
  const float* ly_W3 = (const float*)d_in[49]; const float* ly_b3 = (const float*)d_in[50];
  const float* ne_W = (const float*)d_in[51];  const float* ne_b = (const float*)d_in[52];
  const float* he_W = (const float*)d_in[53];  const float* he_b = (const float*)d_in[54];
  const float* mun_W = (const float*)d_in[55]; const float* mun_b = (const float*)d_in[56];
  const float* mue_W = (const float*)d_in[57]; const float* mue_b = (const float*)d_in[58];

  // workspace bump allocator
  char* wsb = (char*)d_ws;
  auto alloc = [&](size_t nfloats) -> float* {
    float* p = (float*)wsb; wsb += nfloats * sizeof(float); return p;
  };
  float* xn_mi  = alloc((size_t)MI_N * DIN_N);
  float* xn_dis = alloc((size_t)DIS_N * DIN_N);
  float* xc_mi  = alloc((size_t)MI_N * HID_N);   // KAN output, then HGCN running state
  float* xc_dis = alloc((size_t)DIS_N * HID_N);
  float* jk_mi  = alloc((size_t)MI_N * JK_D);
  float* jk_dis = alloc((size_t)DIS_N * JK_D);
  float* tbuf   = alloc((size_t)MI_N * HID_N);
  float* outlin = alloc((size_t)MI_N * HID_N);
  float* g1buf  = alloc((size_t)MI_N * 64);
  float* glog   = alloc((size_t)MI_N * HID_N);
  float* wefft  = alloc((size_t)HID_N * HID_N);
  float* f_mi   = alloc((size_t)MI_N * HID_N);
  float* f_dis  = alloc((size_t)DIS_N * HID_N);
  float* hx1    = alloc((size_t)MI_N * 256);
  float* hx2    = alloc((size_t)MI_N * 128);
  float* xb     = alloc((size_t)MI_N * 64);
  float* hy1    = alloc((size_t)DIS_N * 256);
  float* hy2    = alloc((size_t)DIS_N * 128);
  float* yb     = alloc((size_t)DIS_N * 64);
  float* res_h  = alloc((size_t)MI_N * DIS_N);
  float* znh    = alloc((size_t)MI_N * H1_N);
  float* zeh    = alloc((size_t)DIS_N * H1_N);
  float* znm    = alloc((size_t)MI_N * OUT_N);
  float* zem    = alloc((size_t)DIS_N * OUT_N);

  // ---- FastKAN encoders ----
  auto kan = [&](int M, const float* x, const float* lg, const float* lb,
                 const float* Ws, const float* Wb, const float* bb,
                 float* xn, float* e) {
    ln_rows<<<M, 256, 0, stream>>>(x, lg, lb, xn, DIN_N);
    // base: silu(x) @ Wb.T + bb
    gemm_wmma<1, 0, 0, 1, 0><<<dim3(HID_N / BN, M / BM), 256, 0, stream>>>(
        x, Wb, bb, nullptr, e, M, HID_N, DIN_N, DIN_N, DIN_N, HID_N);
    // spline: basis(xn) @ Ws.T, accumulated
    gemm_wmma<2, 0, 1, 0, 0><<<dim3(HID_N / BN, M / BM), 256, 0, stream>>>(
        xn, Ws, nullptr, rbf_grid, e, M, HID_N, DIN_N * NG_N, DIN_N, DIN_N * NG_N, HID_N);
  };
  kan(MI_N,  concat_mi,  mi_kln_g,  mi_kln_b,  mi_kWs,  mi_kWb,  mi_kbb,  xn_mi,  xc_mi);
  kan(DIS_N, concat_dis, dis_kln_g, dis_kln_b, dis_kWs, dis_kWb, dis_kbb, xn_dis, xc_dis);

  // ---- HGCN stacks ----
  auto hgcn = [&](int M, const float* G, const float* W, const float* attn,
                  const float* bias, const float* rW1, const float* rb1,
                  const float* rW2, const float* rb2, const float* lng,
                  const float* lnb, float* xc, float* jk) {
    for (int l = 0; l < L_N; ++l) {
      weff_kernel<<<256, 256, 0, stream>>>(W + (size_t)l * 8 * HID_N * HID_N,
                                           attn + (size_t)l * 8, wefft);
      // t = xc @ W_eff  (wefft stored [out][in] = [N,K])
      gemm_wmma<0, 0, 0, 0, 0><<<dim3(HID_N / BN, M / BM), 256, 0, stream>>>(
          xc, wefft, nullptr, nullptr, tbuf, M, HID_N, HID_N, HID_N, HID_N, HID_N);
      // out = G @ t + b   (B stored [K,N])
      gemm_wmma<0, 1, 0, 1, 0><<<dim3(HID_N / BN, M / BM), 256, 0, stream>>>(
          G, tbuf, bias + (size_t)l * HID_N, nullptr, outlin, M, HID_N, M, M, HID_N, HID_N);
      // g1 = relu(out @ rW1.T + rb1)
      gemm_wmma<0, 0, 0, 1, 1><<<dim3(1, M / BM), 256, 0, stream>>>(
          outlin, rW1 + (size_t)l * 64 * HID_N, rb1 + (size_t)l * 64, nullptr,
          g1buf, M, 64, HID_N, HID_N, HID_N, 64);
      // glog = g1 @ rW2.T + rb2
      gemm_wmma<0, 0, 0, 1, 0><<<dim3(HID_N / BN, M / BM), 256, 0, stream>>>(
          g1buf, rW2 + (size_t)l * HID_N * 64, rb2 + (size_t)l * HID_N, nullptr,
          glog, M, HID_N, 64, 64, 64, HID_N);
      gate_ln_leaky<<<M, 256, 0, stream>>>(outlin, glog, lng + (size_t)l * HID_N,
                                           lnb + (size_t)l * HID_N, xc,
                                           jk + (size_t)l * HID_N);
    }
  };
  hgcn(MI_N, G_mi, mi_W, mi_attn, mi_bias, mi_rW1, mi_rb1, mi_rW2, mi_rb2,
       mi_lng, mi_lnb, xc_mi, jk_mi);
  gemm_wmma<0, 0, 0, 1, 0><<<dim3(HID_N / BN, MI_N / BM), 256, 0, stream>>>(
      jk_mi, mi_jkW, mi_jkb, nullptr, f_mi, MI_N, HID_N, JK_D, JK_D, JK_D, HID_N);

  hgcn(DIS_N, G_dis, dis_W, dis_attn, dis_bias, dis_rW1, dis_rb1, dis_rW2, dis_rb2,
       dis_lng, dis_lnb, xc_dis, jk_dis);
  gemm_wmma<0, 0, 0, 1, 0><<<dim3(HID_N / BN, DIS_N / BM), 256, 0, stream>>>(
      jk_dis, dis_jkW, dis_jkb, nullptr, f_dis, DIS_N, HID_N, JK_D, JK_D, JK_D, HID_N);

  // ---- MLP heads ----
  auto mlp = [&](int M, const float* f, const float* W1, const float* b1,
                 const float* W2, const float* b2, const float* W3, const float* b3,
                 float* h1p, float* h2p, float* o) {
    gemm_wmma<0, 0, 0, 1, 1><<<dim3(256 / BN, M / BM), 256, 0, stream>>>(
        f, W1, b1, nullptr, h1p, M, 256, HID_N, HID_N, HID_N, 256);
    gemm_wmma<0, 0, 0, 1, 1><<<dim3(128 / BN, M / BM), 256, 0, stream>>>(
        h1p, W2, b2, nullptr, h2p, M, 128, 256, 256, 256, 128);
    gemm_wmma<0, 0, 0, 1, 1><<<dim3(1, M / BM), 256, 0, stream>>>(
        h2p, W3, b3, nullptr, o, M, 64, 128, 128, 128, 64);
  };
  mlp(MI_N,  f_mi,  lx_W1, lx_b1, lx_W2, lx_b2, lx_W3, lx_b3, hx1, hx2, xb);
  mlp(DIS_N, f_dis, ly_W1, ly_b1, ly_W2, ly_b2, ly_W3, ly_b3, hy1, hy2, yb);

  // result_h = x @ y.T
  gemm_wmma<0, 0, 0, 0, 0><<<dim3(DIS_N / BN, MI_N / BM), 256, 0, stream>>>(
      xb, yb, nullptr, nullptr, res_h, MI_N, DIS_N, 64, 64, 64, DIS_N);

  // ---- VAE-like branch ----
  gemm_wmma<0, 0, 0, 1, 1><<<dim3(H1_N / BN, MI_N / BM), 256, 0, stream>>>(
      concat_mi, ne_W, ne_b, nullptr, znh, MI_N, H1_N, DIN_N, DIN_N, DIN_N, H1_N);
  gemm_wmma<0, 0, 0, 1, 1><<<dim3(H1_N / BN, DIS_N / BM), 256, 0, stream>>>(
      concat_dis, he_W, he_b, nullptr, zeh, DIS_N, H1_N, DIN_N, DIN_N, DIN_N, H1_N);
  gemm_wmma<0, 0, 0, 1, 0><<<dim3(1, MI_N / BM), 256, 0, stream>>>(
      znh, mun_W, mun_b, nullptr, znm, MI_N, OUT_N, H1_N, H1_N, H1_N, OUT_N);
  gemm_wmma<0, 0, 0, 1, 0><<<dim3(1, DIS_N / BM), 256, 0, stream>>>(
      zeh, mue_W, mue_b, nullptr, zem, DIS_N, OUT_N, H1_N, H1_N, H1_N, OUT_N);

  float* score = (float*)d_out;
  float* recon = score + (size_t)MI_N * DIS_N;
  gemm_wmma<0, 0, 0, 0, 0><<<dim3(DIS_N / BN, MI_N / BM), 256, 0, stream>>>(
      znm, zem, nullptr, nullptr, recon, MI_N, DIS_N, OUT_N, OUT_N, OUT_N, DIS_N);

  score_kernel<<<(MI_N * DIS_N) / 256, 256, 0, stream>>>(score, recon, res_h,
                                                         MI_N * DIS_N);
}
